// NerfModel_25142738551283
// MI455X (gfx1250) — matrix operand order repaired and checked by reading, not compiled, not run on hardware
//
#include <hip/hip_runtime.h>
#include <hip/hip_bf16.h>

#define BLOCK 128
#define WAVES 4
#define RS 328      // activation row stride (halves): 16B * 41 -> conflict-free b128 reads
#define WS 40       // staged-weight row stride (halves): 80B = 4*5 dwords -> conflict-free
#define ZHALVES (WAVES * 16 * RS)   // 20992 halves
#define WCHALVES (256 * WS)         // 10240 halves

typedef _Float16 v16h __attribute__((ext_vector_type(16)));
typedef _Float16 v8h  __attribute__((ext_vector_type(8)));
typedef float    v8f  __attribute__((ext_vector_type(8)));
typedef unsigned int uint32x4 __attribute__((ext_vector_type(4)));
typedef int int32x8 __attribute__((ext_vector_type(8)));
typedef int int32x4 __attribute__((ext_vector_type(4)));

// ---- TDM availability (probe-verified builtins; arity differs per toolchain) ----
#if defined(__HIP_DEVICE_COMPILE__) && defined(__gfx1250__)
#if __has_builtin(__builtin_amdgcn_tensor_load_to_lds) && __has_builtin(__builtin_amdgcn_s_wait_tensorcnt)
#define TDM_OK 1
#endif
#endif

// Prepped-weight layout in d_ws (halves): per layer, per 32-K chunk, the exact
// LDS image: NPAD rows of WS halves (transposed f16, zero-padded).
__host__ __device__ constexpr size_t pre_off(int layer) {
  // sizes: L0:2*256*40, L1..L3:8*256*40, L4:10*256*40, L5..L7:8*256*40, L8:9*128*40, L9:4*16*40
  const size_t s[10] = {20480, 81920, 81920, 81920, 102400, 81920, 81920, 81920, 46080, 2560};
  size_t o = 0;
  for (int i = 0; i < layer; ++i) o += s[i];
  return o;
}
#define PRE_TOTAL_HALVES 663040ull

__device__ __forceinline__ v16h cat8(v8h a, v8h b) {
  return __builtin_shufflevector(a, b, 0,1,2,3,4,5,6,7,8,9,10,11,12,13,14,15);
}

#if defined(TDM_OK)
// 1-D contiguous global->LDS DMA via the Tensor Data Mover.
// D# group0: count=1, lds_addr, 57b global_addr, type=2. group1: data_size=8B,
// tensor_dim0 = tile_dim0 = n8 elements, stride = n8, everything else 0.
__device__ __forceinline__ void tdm_copy_1d(const void* gsrc, unsigned ldsByteOff, unsigned bytes) {
  const unsigned long long ga = (unsigned long long)gsrc;
  const unsigned n8 = bytes >> 3;                       // 8-byte elements
  uint32x4 g0;
  g0[0] = 1u;                                           // count=1
  g0[1] = ldsByteOff;                                   // lds_addr [63:32]
  g0[2] = (unsigned)(ga & 0xFFFFFFFFu);                 // global_addr [95:64]
  g0[3] = (unsigned)((ga >> 32) & 0x1FFFFFFu) | (2u << 30); // ga[56:32] + type=2
  int32x8 g1 = {0, 0, 0, 0, 0, 0, 0, 0};
  g1[0] = (3 << 16);                                    // data_size=3 (8B), wg_mask=0
  g1[1] = (int)((n8 & 0xFFFFu) << 16);                  // tensor_dim0[15:0]  @ [63:48]
  g1[2] = (int)(n8 >> 16);                              // tensor_dim0[31:16] @ [79:64]
  g1[3] = (int)((n8 & 0xFFFFu) << 16);                  // tile_dim0          @ [127:112]
  g1[5] = (int)n8;                                      // tensor_dim0_stride low32
  int32x4 z4 = {0, 0, 0, 0};
#if __clang_major__ >= 23
  int32x8 z8 = {0, 0, 0, 0, 0, 0, 0, 0};
  __builtin_amdgcn_tensor_load_to_lds(g0, g1, z4, z4, z8, 0);
#else
  __builtin_amdgcn_tensor_load_to_lds(g0, g1, z4, z4, 0);
#endif
}
#endif

// ---- one-shot weight prep: f32 [KDIM][NREAL] -> f16 transposed chunked LDS image
template<int KDIM, int NREAL, int NPAD>
__global__ void prep_weights(const float* __restrict__ W, _Float16* __restrict__ dst) {
  constexpr int KC = ((KDIM + 31) & ~31) / 32;
  const int total = KC * NPAD * WS;
  for (int idx = blockIdx.x * blockDim.x + threadIdx.x; idx < total;
       idx += gridDim.x * blockDim.x) {
    const int kc = idx / (NPAD * WS);
    const int r  = idx - kc * (NPAD * WS);
    const int n  = r / WS;
    const int kk = r - n * WS;
    const int k  = kc * 32 + kk;
    float v = 0.0f;
    if (kk < 32 && k < KDIM && n < NREAL) v = W[k * NREAL + n];
    dst[idx] = (_Float16)v;
  }
}

// One MLP layer: OUT[16 x NREAL] = act(IN[16 x KDIM] @ W[KDIM x NREAL] + b)
template<int KDIM, int NREAL, int NPAD, bool RELU, bool FINAL>
__device__ __forceinline__ void mlp_layer(
    const float* __restrict__ W, const _Float16* __restrict__ Wpre,
    const float* __restrict__ bias,
    _Float16* zrow, _Float16* Wc, int tid, int lane, int wid,
    float* __restrict__ out, int N, int rayBase)
{
  constexpr int KPAD = (KDIM + 31) & ~31;
  constexpr int KC   = KPAD / 32;
  constexpr int NT   = NPAD / 16;
  const int n0 = lane & 15;        // A row (M) and B/D column (N) sub-index
  const int hi = lane >> 4;

  v8f acc[NT];
  #pragma unroll
  for (int nt = 0; nt < NT; ++nt) {
    const int n = nt * 16 + n0;
    const float bv = (n < NREAL) ? bias[n] : 0.0f;
    #pragma unroll
    for (int i = 0; i < 8; ++i) acc[nt][i] = bv;
  }

  // A fragment loader (16x32 f16): halves e0..7 -> K=kb+8*hi+e, e8..15 -> K=kb+16+8*hi+(e-8)
  auto loadA = [&](int kb) -> v16h {
    const _Float16* ap = zrow + n0 * RS + kb + 8 * hi;
    return cat8(*(const v8h*)ap, *(const v8h*)(ap + 16));
  };
  // B fragment loader (32x16 f16): halves e0..7 -> K=8*hi+e, e8..15 -> K=16+8*hi+(e-8), column n = nt*16+n0
  auto loadB = [&](int nt) -> v16h {
    const _Float16* bq = Wc + (nt * 16 + n0) * WS + 8 * hi;
    return cat8(*(const v8h*)bq, *(const v8h*)(bq + 16));
  };

  v16h a = loadA(0);

  #pragma unroll 1
  for (int kc = 0; kc < KC; ++kc) {
    const int kbase = kc * 32;

    __syncthreads();   // all waves done reading previous Wc chunk
    if (Wpre) {
#if defined(TDM_OK)
      if (wid == 0) {  // one DMA per chunk; contiguous 8B-aligned copy
        tdm_copy_1d(Wpre + (size_t)kc * (NPAD * WS),
                    (unsigned)(unsigned long long)(void*)Wc,
                    (unsigned)(NPAD * WS * 2));
        __builtin_amdgcn_s_wait_tensorcnt((short)0);
      }
#else
      const _Float16* src = Wpre + (size_t)kc * (NPAD * WS);
      #pragma unroll 2
      for (int idx = tid * 8; idx < NPAD * WS; idx += BLOCK * 8)
        *(v8h*)(Wc + idx) = *(const v8h*)(src + idx);
#endif
    } else {
      // fallback: stage from f32 weights, transpose + convert on the fly
      #pragma unroll 4
      for (int idx = tid; idx < NPAD * 32; idx += BLOCK) {
        const int kk = idx / NPAD;
        const int n  = idx - kk * NPAD;
        const int k  = kbase + kk;
        float v = 0.0f;
        if (k < KDIM && n < NREAL) v = W[k * NREAL + n];
        Wc[n * WS + kk] = (_Float16)v;
      }
    }
    __syncthreads();   // chunk visible to all waves

    // prefetch next chunk's A fragment (input cols are stable all layer)
    const v16h an = (kc + 1 < KC) ? loadA(kbase + 32) : a;

    // B fragments software-pipelined at depth 3: each fragment is issued two
    // WMMAs ahead of its use -> partial s_wait_dscnt, LDS latency fully hidden
    v16h b0 = loadB(0);
    v16h b1 = (NT > 1) ? loadB(1) : b0;
    #pragma unroll
    for (int nt = 0; nt < NT; ++nt) {
      const v16h bn = (nt + 2 < NT) ? loadB(nt + 2) : b0;
      acc[nt] = __builtin_amdgcn_wmma_f32_16x16x32_f16(
          false, a, false, b0, (short)0, acc[nt], false, false);
      b0 = b1;
      b1 = bn;
    }
    a = an;
  }

  if (!FINAL) {
    // D layout: VGPR r holds (M = r + 8*hi, N = nt*16 + n0)
    #pragma unroll
    for (int nt = 0; nt < NT; ++nt) {
      const int n = nt * 16 + n0;
      #pragma unroll
      for (int r = 0; r < 8; ++r) {
        float v = acc[nt][r];
        if (RELU) v = fmaxf(v, 0.0f);
        zrow[(r + 8 * hi) * RS + n] = (_Float16)v;
      }
    }
  } else {
    // NT == 1, only columns 0..3 real: 3=density(softplus), 0..2=rgb(sigmoid)
    #pragma unroll
    for (int r = 0; r < 8; ++r) {
      const int ray = rayBase + r + 8 * hi;
      if (ray < N) {
        const float v = acc[0][r];
        if (n0 == 3) {
          out[ray] = (v > 8.0f) ? v : log1pf(__expf(v));
        } else if (n0 < 3) {
          out[N + ray * 3 + n0] = 1.0f / (1.0f + __expf(-v));
        }
      }
    }
  }
}

__global__ __launch_bounds__(BLOCK)
void nerf_fused_kernel(
    const float* __restrict__ pos, const float* __restrict__ dir,
    const float* w00, const float* b00, const float* w01, const float* b01,
    const float* w02, const float* b02, const float* w03, const float* b03,
    const float* w10, const float* b10, const float* w11, const float* b11,
    const float* w12, const float* b12, const float* w13, const float* b13,
    const float* cw0, const float* cb0, const float* cw1, const float* cb1,
    const _Float16* __restrict__ pre,   // prepped weights in d_ws (or null)
    float* __restrict__ out, int N)
{
  __shared__ __align__(16) _Float16 smem[ZHALVES + WCHALVES];   // 62464 B

  const int tid  = threadIdx.x;
  const int lane = tid & 31;
  const int wid  = tid >> 5;
  _Float16* zrow = smem + wid * (16 * RS);   // this wave's private 16-ray tile
  _Float16* Wc   = smem + ZHALVES;           // shared staged-weight chunk
  const int rayBase = blockIdx.x * (WAVES * 16) + wid * 16;

  const int m  = lane & 15;
  const int hi = lane >> 4;

  // ---- positional encoding: 63 comps (+1 zero pad) -> cols [0..63] and [256..319]
  {
    const int ray = rayBase + m;
    const int rc  = (ray < N) ? ray : (N - 1);
    const float p0 = pos[rc * 3 + 0];
    const float p1 = pos[rc * 3 + 1];
    const float p2 = pos[rc * 3 + 2];
    #pragma unroll
    for (int cc = 0; cc < 32; ++cc) {
      const int c = hi * 32 + cc;
      float v;
      if (c < 3) {
        v = (c == 0) ? p0 : ((c == 1) ? p1 : p2);
      } else if (c < 33) {
        const int j = c - 3;  const int coord = j / 10;
        const float x = (coord == 0) ? p0 : ((coord == 1) ? p1 : p2);
        v = __sinf(x * (float)(1 << (j % 10)));
      } else if (c < 63) {
        const int j = c - 33; const int coord = j / 10;
        const float x = (coord == 0) ? p0 : ((coord == 1) ? p1 : p2);
        v = __cosf(x * (float)(1 << (j % 10)));
      } else {
        v = 0.0f;
      }
      const _Float16 h = (_Float16)v;
      zrow[m * RS + c]       = h;   // layer-0 input
      zrow[m * RS + 256 + c] = h;   // persistent copy for the w10 concat
    }
  }

  const _Float16* P0 = pre ? pre + pre_off(0) : nullptr;
  const _Float16* P1 = pre ? pre + pre_off(1) : nullptr;
  const _Float16* P2 = pre ? pre + pre_off(2) : nullptr;
  const _Float16* P3 = pre ? pre + pre_off(3) : nullptr;
  const _Float16* P4 = pre ? pre + pre_off(4) : nullptr;
  const _Float16* P5 = pre ? pre + pre_off(5) : nullptr;
  const _Float16* P6 = pre ? pre + pre_off(6) : nullptr;
  const _Float16* P7 = pre ? pre + pre_off(7) : nullptr;
  const _Float16* P8 = pre ? pre + pre_off(8) : nullptr;
  const _Float16* P9 = pre ? pre + pre_off(9) : nullptr;

  // ---- density trunk
  mlp_layer< 63, 256, 256, true , false>(w00, P0, b00, zrow, Wc, tid, lane, wid, out, N, rayBase);
  mlp_layer<256, 256, 256, true , false>(w01, P1, b01, zrow, Wc, tid, lane, wid, out, N, rayBase);
  mlp_layer<256, 256, 256, true , false>(w02, P2, b02, zrow, Wc, tid, lane, wid, out, N, rayBase);
  mlp_layer<256, 256, 256, true , false>(w03, P3, b03, zrow, Wc, tid, lane, wid, out, N, rayBase);
  mlp_layer<319, 256, 256, true , false>(w10, P4, b10, zrow, Wc, tid, lane, wid, out, N, rayBase); // concat(z, pe)
  mlp_layer<256, 256, 256, true , false>(w11, P5, b11, zrow, Wc, tid, lane, wid, out, N, rayBase);
  mlp_layer<256, 256, 256, true , false>(w12, P6, b12, zrow, Wc, tid, lane, wid, out, N, rayBase);
  mlp_layer<256, 256, 256, false, false>(w13, P7, b13, zrow, Wc, tid, lane, wid, out, N, rayBase); // no relu

  // ---- direction encoding: 27 comps (+5 zero pad) overwrite cols [256..287]
  // (per-wave private rows; LDS ops are in-order within a wave -> no barrier needed)
  {
    const int ray = rayBase + m;
    const int rc  = (ray < N) ? ray : (N - 1);
    const float q0 = dir[rc * 3 + 0];
    const float q1 = dir[rc * 3 + 1];
    const float q2 = dir[rc * 3 + 2];
    #pragma unroll
    for (int cc = 0; cc < 16; ++cc) {
      const int c = hi * 16 + cc;
      float v;
      if (c < 3) {
        v = (c == 0) ? q0 : ((c == 1) ? q1 : q2);
      } else if (c < 15) {
        const int j = c - 3;  const int coord = j / 4;
        const float x = (coord == 0) ? q0 : ((coord == 1) ? q1 : q2);
        v = __sinf(x * (float)(1 << (j % 4)));
      } else if (c < 27) {
        const int j = c - 15; const int coord = j / 4;
        const float x = (coord == 0) ? q0 : ((coord == 1) ? q1 : q2);
        v = __cosf(x * (float)(1 << (j % 4)));
      } else {
        v = 0.0f;
      }
      zrow[m * RS + 256 + c] = (_Float16)v;
    }
  }

  // ---- color head
  mlp_layer<283, 128, 128, true , false>(cw0, P8, cb0, zrow, Wc, tid, lane, wid, out, N, rayBase); // concat(z, de)
  mlp_layer<128,   4,  16, false, true >(cw1, P9, cb1, zrow, Wc, tid, lane, wid, out, N, rayBase); // final
}

extern "C" void kernel_launch(void* const* d_in, const int* in_sizes, int n_in,
                              void* d_out, int out_size, void* d_ws, size_t ws_size,
                              hipStream_t stream) {
  const float* pos = (const float*)d_in[0];
  const float* dir = (const float*)d_in[1];
  // d_in[2..4] = pixel_width, t_start, t_end (unused by the reference math)
  const float* w00 = (const float*)d_in[5];  const float* b00 = (const float*)d_in[6];
  const float* w01 = (const float*)d_in[7];  const float* b01 = (const float*)d_in[8];
  const float* w02 = (const float*)d_in[9];  const float* b02 = (const float*)d_in[10];
  const float* w03 = (const float*)d_in[11]; const float* b03 = (const float*)d_in[12];
  const float* w10 = (const float*)d_in[13]; const float* b10 = (const float*)d_in[14];
  const float* w11 = (const float*)d_in[15]; const float* b11 = (const float*)d_in[16];
  const float* w12 = (const float*)d_in[17]; const float* b12 = (const float*)d_in[18];
  const float* w13 = (const float*)d_in[19]; const float* b13 = (const float*)d_in[20];
  const float* cw0 = (const float*)d_in[21]; const float* cb0 = (const float*)d_in[22];
  const float* cw1 = (const float*)d_in[23]; const float* cb1 = (const float*)d_in[24];

  const int N = in_sizes[0] / 3;                 // 262144
  const int grid = (N + WAVES * 16 - 1) / (WAVES * 16);

  // Prep f16 transposed weights into d_ws if it is large enough (deterministic:
  // same decision & same work every call).
  _Float16* pre = (ws_size >= PRE_TOTAL_HALVES * 2) ? (_Float16*)d_ws : nullptr;
  if (pre) {
    prep_weights< 63, 256, 256><<<80, 256, 0, stream>>>(w00, pre + pre_off(0));
    prep_weights<256, 256, 256><<<320, 256, 0, stream>>>(w01, pre + pre_off(1));
    prep_weights<256, 256, 256><<<320, 256, 0, stream>>>(w02, pre + pre_off(2));
    prep_weights<256, 256, 256><<<320, 256, 0, stream>>>(w03, pre + pre_off(3));
    prep_weights<319, 256, 256><<<400, 256, 0, stream>>>(w10, pre + pre_off(4));
    prep_weights<256, 256, 256><<<320, 256, 0, stream>>>(w11, pre + pre_off(5));
    prep_weights<256, 256, 256><<<320, 256, 0, stream>>>(w12, pre + pre_off(6));
    prep_weights<256, 256, 256><<<320, 256, 0, stream>>>(w13, pre + pre_off(7));
    prep_weights<283, 128, 128><<<180, 256, 0, stream>>>(cw0, pre + pre_off(8));
    prep_weights<128,   4,  16><<<10,  256, 0, stream>>>(cw1, pre + pre_off(9));
  }

  nerf_fused_kernel<<<grid, BLOCK, 0, stream>>>(
      pos, dir,
      w00, b00, w01, b01, w02, b02, w03, b03,
      w10, b10, w11, b11, w12, b12, w13, b13,
      cw0, cb0, cw1, cb1,
      pre, (float*)d_out, N);
}
